// TrueVectorizedSlidingWindowOuterLoopModel_4002909520750
// MI455X (gfx1250) — compile-verified, matrix operand order checked by reading, not run on hardware
//
#include <hip/hip_runtime.h>

// ---------------------------------------------------------------------------
// Problem constants (B=1): x[2048,512]; 4 weights [512,512]; window 64.
// The attention middle is CHANNEL-WISE (softmax over the window per channel),
// so only the 4 projections are matrix ops -> bf16 WMMA GEMMs.
// ---------------------------------------------------------------------------
constexpr int S   = 2048;
constexpr int DM  = 512;
constexpr int WIN = 64;   // window size
// head dim = 32 -> scale = 1/sqrt(32)
#define ATTN_SCALE 0.1767766952966368811f

typedef __bf16 bf16_t;
typedef __attribute__((ext_vector_type(16))) __bf16 v16bf;
typedef __attribute__((ext_vector_type(8)))  float  v8f;
typedef __attribute__((ext_vector_type(4)))  int    v4i;

union Frag16 { v16bf v; v4i q[2]; };

// ---------------------------------------------------------------------------
// f32 -> bf16 conversion (x and weights), RNE via clang __bf16 cast.
// ---------------------------------------------------------------------------
__global__ __launch_bounds__(256) void f32_to_bf16_kernel(
    const float* __restrict__ in, bf16_t* __restrict__ out, int n) {
  int i = blockIdx.x * blockDim.x + threadIdx.x;
  if (i < n) out[i] = (bf16_t)in[i];
}

// ---------------------------------------------------------------------------
// NT GEMM:  C[m,n] = sum_k A[m,k] * B[n,k]   (A: S x DM, B: DM x DM, C: f32)
//
// Block = 8 waves = 128x64 C tile; wave = 16x64 strip (4 accumulators).
// B tile (64 rows x 512 K = 64 KB bf16) is staged ONCE per block into LDS via
// the gfx1250 async-to-LDS path (global_load_async_to_lds_b128, ASYNCcnt),
// removing the 8x intra-block B redundancy. LDS rows padded to 520 elements
// (1040 B) so half-wave fragment reads stripe conflict-free over 64 banks.
// A fragments stream from global (L2-resident) with one-step double buffering
// so next-step loads overlap the current 4 WMMAs.
//
// Fragment layout per CDNA5 ISA (16-bit A 16x32): lane L holds row (L&15),
// K slots { 8*(L>>4) .. +7 } and { 16 + 8*(L>>4) .. +7 } -> two contiguous
// 16-byte runs. B (32x16, K x N) mirrors with N = (L&15).
// C layout: VGPR r -> (m = r + 8*(L>>4), n = L&15).
// ---------------------------------------------------------------------------
constexpr int LDB = 520;   // padded LDS row stride (elements)

__global__ __launch_bounds__(256) void gemm_nt_wmma_kernel(
    const bf16_t* __restrict__ A,   // [S,  DM] bf16 (row-major)
    const bf16_t* __restrict__ B,   // [DM, DM] bf16 (row-major; we use B^T)
    float* __restrict__ C) {        // [S,  DM] f32
  __shared__ __align__(16) bf16_t ldsB[64 * LDB];   // ~66.6 KB of 320 KB WGP LDS

  const int lane = threadIdx.x & 31;
  const int wave = threadIdx.x >> 5;
  const int half = lane >> 4;      // 0 or 1
  const int r16  = lane & 15;

  const int mBase = blockIdx.x * 128 + wave * 16;
  const int nBase = blockIdx.y * 64;

  // --- async-stage the 64x512 B tile into LDS (4096 x 16-byte chunks) ------
  for (int c = threadIdx.x; c < 64 * 64; c += 256) {
    const int row = c >> 6;                 // 0..63   (64 chunks per row)
    const int col = (c & 63) << 3;          // element offset, 8 bf16 per 16 B
    const bf16_t* g = B + (size_t)(nBase + row) * DM + col;
    const unsigned int loff =
        (unsigned int)(uintptr_t)(&ldsB[row * LDB + col]);  // LDS byte offset
    asm volatile("global_load_async_to_lds_b128 %0, %1, off"
                 :
                 : "v"(loff), "v"((unsigned long long)(uintptr_t)g)
                 : "memory");
  }
  asm volatile("s_wait_asynccnt 0x0" ::: "memory");
  __syncthreads();

  const bf16_t* aRow = A + (size_t)(mBase + r16) * DM;
  const bf16_t* lb0  = &ldsB[( 0 + r16) * LDB];
  const bf16_t* lb1  = &ldsB[(16 + r16) * LDB];
  const bf16_t* lb2  = &ldsB[(32 + r16) * LDB];
  const bf16_t* lb3  = &ldsB[(48 + r16) * LDB];

  v8f acc0 = {}, acc1 = {}, acc2 = {}, acc3 = {};

  // --- K loop with one-step A double buffering -----------------------------
  Frag16 aCur, aNxt;
  {
    const int o0 = 8 * half, o1 = 16 + 8 * half;
    aCur.q[0] = *(const v4i*)(aRow + o0);
    aCur.q[1] = *(const v4i*)(aRow + o1);
  }

  for (int kk = 0; kk < DM; kk += 32) {
    const int o0 = kk + 8 * half;        // first 8-elem run
    const int o1 = kk + 16 + 8 * half;   // second 8-elem run

    if (kk + 32 < DM) {                  // prefetch next A fragment
      aNxt.q[0] = *(const v4i*)(aRow + o0 + 32);
      aNxt.q[1] = *(const v4i*)(aRow + o1 + 32);
    }

    Frag16 b0, b1, b2, b3;               // ds_load_b128 x8 from staged tile
    b0.q[0] = *(const v4i*)(lb0 + o0);  b0.q[1] = *(const v4i*)(lb0 + o1);
    b1.q[0] = *(const v4i*)(lb1 + o0);  b1.q[1] = *(const v4i*)(lb1 + o1);
    b2.q[0] = *(const v4i*)(lb2 + o0);  b2.q[1] = *(const v4i*)(lb2 + o1);
    b3.q[0] = *(const v4i*)(lb3 + o0);  b3.q[1] = *(const v4i*)(lb3 + o1);

    acc0 = __builtin_amdgcn_wmma_f32_16x16x32_bf16(false, aCur.v, false, b0.v,
                                                   (short)0, acc0, false, false);
    acc1 = __builtin_amdgcn_wmma_f32_16x16x32_bf16(false, aCur.v, false, b1.v,
                                                   (short)0, acc1, false, false);
    acc2 = __builtin_amdgcn_wmma_f32_16x16x32_bf16(false, aCur.v, false, b2.v,
                                                   (short)0, acc2, false, false);
    acc3 = __builtin_amdgcn_wmma_f32_16x16x32_bf16(false, aCur.v, false, b3.v,
                                                   (short)0, acc3, false, false);
    aCur = aNxt;
  }

#pragma unroll
  for (int r = 0; r < 8; ++r) {
    const size_t m = (size_t)(mBase + r + 8 * half);
    float* cRow = C + m * DM + nBase + r16;
    cRow[ 0] = acc0[r];
    cRow[16] = acc1[r];
    cRow[32] = acc2[r];
    cRow[48] = acc3[r];
  }
}

// ---------------------------------------------------------------------------
// Channel-wise sliding-window softmax-attention (the reference keeps the D
// axis: softmax over the 64-key window independently per channel).
// One thread per (pos, channel); 32 consecutive channels per wave -> coalesced.
// Online (flash-style) single pass; emits ctx directly as bf16 for the O GEMM.
// ---------------------------------------------------------------------------
__global__ __launch_bounds__(256) void swin_channel_softmax_kernel(
    const float* __restrict__ q,
    const float* __restrict__ k,
    const float* __restrict__ v,
    bf16_t* __restrict__ ctx) {
  const int t   = blockIdx.x * blockDim.x + threadIdx.x;
  const int pos = t / DM;
  const int dm  = t - pos * DM;

  const float qd = q[(size_t)pos * DM + dm] * ATTN_SCALE;

  float m = -__builtin_inff();
  float l = 0.0f;
  float acc = 0.0f;

  int j0 = pos - (WIN - 1);
  if (j0 < 0) j0 = 0;

  for (int j = j0; j <= pos; ++j) {
    const float kv = k[(size_t)j * DM + dm];
    const float vv = v[(size_t)j * DM + dm];
    const float sc = qd * kv;
    const float mn = fmaxf(m, sc);
    const float corr = __expf(m - mn);   // first iter: exp(-inf) = 0
    const float p    = __expf(sc - mn);
    l   = l * corr + p;
    acc = acc * corr + p * vv;
    m = mn;
  }

  ctx[(size_t)pos * DM + dm] = (bf16_t)(acc / l);
}

// ---------------------------------------------------------------------------
// Host-side orchestration (all launches on `stream`; no allocs; deterministic)
// ---------------------------------------------------------------------------
extern "C" void kernel_launch(void* const* d_in, const int* in_sizes, int n_in,
                              void* d_out, int out_size, void* d_ws, size_t ws_size,
                              hipStream_t stream) {
  (void)in_sizes; (void)n_in; (void)out_size; (void)ws_size;

  const float* x  = (const float*)d_in[0];
  const float* Wq = (const float*)d_in[1];
  const float* Wk = (const float*)d_in[2];
  const float* Wv = (const float*)d_in[3];
  const float* Wo = (const float*)d_in[4];
  float* out = (float*)d_out;

  constexpr size_t XB = (size_t)S * DM;    // 1,048,576 elems
  constexpr size_t WN = (size_t)DM * DM;   //   262,144 elems
  constexpr size_t MB = 1024u * 1024u;

  char* ws = (char*)d_ws;
  bf16_t* xbf  = (bf16_t*)(ws + 0);              // 2 MB
  bf16_t* wqb  = (bf16_t*)(ws + 2 * MB);         // 512 KB
  bf16_t* wkb  = (bf16_t*)(ws + 2 * MB + 512 * 1024);
  bf16_t* wvb  = (bf16_t*)(ws + 3 * MB);
  bf16_t* wob  = (bf16_t*)(ws + 3 * MB + 512 * 1024);
  float*  qf   = (float*)(ws + 4 * MB);          // 4 MB
  float*  kf   = (float*)(ws + 8 * MB);          // 4 MB
  float*  vf   = (float*)(ws + 12 * MB);         // 4 MB
  bf16_t* ctxb = (bf16_t*)(ws + 16 * MB);        // 2 MB   (total 18 MB)

  // 1) precision conversion
  f32_to_bf16_kernel<<<(int)((XB + 255) / 256), 256, 0, stream>>>(x,  xbf, (int)XB);
  f32_to_bf16_kernel<<<(int)((WN + 255) / 256), 256, 0, stream>>>(Wq, wqb, (int)WN);
  f32_to_bf16_kernel<<<(int)((WN + 255) / 256), 256, 0, stream>>>(Wk, wkb, (int)WN);
  f32_to_bf16_kernel<<<(int)((WN + 255) / 256), 256, 0, stream>>>(Wv, wvb, (int)WN);
  f32_to_bf16_kernel<<<(int)((WN + 255) / 256), 256, 0, stream>>>(Wo, wob, (int)WN);

  // 2) Q/K/V projections (x @ W.T) via bf16 WMMA, f32 results
  dim3 gGemm(S / 128, DM / 64);   // (16, 8)
  gemm_nt_wmma_kernel<<<gGemm, 256, 0, stream>>>(xbf, wqb, qf);
  gemm_nt_wmma_kernel<<<gGemm, 256, 0, stream>>>(xbf, wkb, kf);
  gemm_nt_wmma_kernel<<<gGemm, 256, 0, stream>>>(xbf, wvb, vf);

  // 3) channel-wise sliding-window softmax attention -> bf16 context
  swin_channel_softmax_kernel<<<(int)(XB / 256), 256, 0, stream>>>(qf, kf, vf, ctxb);

  // 4) output projection (ctx @ Wo.T) -> f32 d_out
  gemm_nt_wmma_kernel<<<gGemm, 256, 0, stream>>>(ctxb, wob, out);
}
